// MHDPARelationModule_75239237091645
// MI455X (gfx1250) — matrix operand order repaired and checked by reading, not compiled, not run on hardware
//
#include <hip/hip_runtime.h>

// ---------------- problem constants ----------------
#define NB   16      // batch
#define ND   64      // depth channels
#define NXY  48      // spatial
#define NN   2304    // 48*48 entities
#define DIN  66      // ND + 2 coords
#define DINP 96      // padded inner dim (3 * 32)
#define DI   128     // interaction dim
#define NT   144     // NN / 16

typedef __attribute__((ext_vector_type(16))) __bf16 bf16x16;
typedef __attribute__((ext_vector_type(8)))  float  f32x8;

union Frag {
    bf16x16 v;
    uint4   q[2];
    unsigned u[8];
};

static __device__ __forceinline__ unsigned short f2bf(float f) {
    unsigned x = __float_as_uint(f);
    x += 0x7FFFu + ((x >> 16) & 1u);       // round-to-nearest-even
    return (unsigned short)(x >> 16);
}
static __device__ __forceinline__ unsigned pk(float lo, float hi) {
    return (unsigned)f2bf(lo) | ((unsigned)f2bf(hi) << 16);
}
static __device__ __forceinline__ f32x8 wmma_bf16(bf16x16 a, bf16x16 b, f32x8 c) {
    return __builtin_amdgcn_wmma_f32_16x16x32_bf16(false, a, false, b,
                                                   (short)0, c, false, false);
}

// ---------------- kernel 0: pad + convert weights to bf16, fold scale into Wq
__global__ void prep_w(const float* __restrict__ Wq, const float* __restrict__ Wk,
                       const float* __restrict__ Wv, unsigned short* __restrict__ Wpad) {
    int idx = blockIdx.x * blockDim.x + threadIdx.x;
    if (idx >= 3 * DI * DINP) return;
    int mi  = idx / (DI * DINP);
    int rem = idx % (DI * DINP);
    int o = rem / DINP, c = rem % DINP;
    const float* W = (mi == 0) ? Wq : (mi == 1) ? Wk : Wv;
    float v = (c < DIN) ? W[o * DIN + c] : 0.0f;
    if (mi == 0) v *= 0.08838834764831845f;   // 1/sqrt(DI), folded into Q
    Wpad[idx] = f2bf(v);
}

// ---------------- kernel 1: QKV projection via WMMA (E built on the fly)
__global__ void __launch_bounds__(32)
qkv_proj(const float* __restrict__ x, const unsigned short* __restrict__ Wpad,
         unsigned short* __restrict__ Qb, unsigned short* __restrict__ Kb,
         unsigned short* __restrict__ Vtb) {
    int b = blockIdx.y, nt = blockIdx.x;
    int lane = threadIdx.x, hi = lane >> 4, lr = lane & 15;
    int n0 = nt * 16, n = n0 + lr;
    const float* xb = x + (size_t)b * ND * NN;

    // A-fragments of E (16 rows x 32 inner) for 3 inner chunks
    Frag ef[3];
#pragma unroll
    for (int c = 0; c < 2; ++c) {
        int dinb = 32 * c + hi * 8;
#pragma unroll
        for (int i = 0; i < 4; ++i) {
            ef[c].u[i]     = pk(xb[(dinb + 2*i)      * NN + n], xb[(dinb + 2*i + 1)  * NN + n]);
            ef[c].u[4 + i] = pk(xb[(dinb + 16 + 2*i) * NN + n], xb[(dinb + 17 + 2*i) * NN + n]);
        }
    }
    // chunk 2: din 64 = fx, din 65 = fy, rest zero padding
    float fx = -1.0f + ((float)(n / NXY) + 0.5f) * (2.0f / NXY);
    float fy = -1.0f + ((float)(n % NXY) + 0.5f) * (2.0f / NXY);
#pragma unroll
    for (int i = 0; i < 8; ++i) ef[2].u[i] = 0u;
    ef[2].u[0] = hi ? 0u : pk(fx, fy);

    f32x8 zf = {};
#pragma unroll
    for (int mi = 0; mi < 3; ++mi) {
        const unsigned short* Wm = Wpad + mi * DI * DINP;
#pragma unroll
        for (int t = 0; t < 8; ++t) {
            int d0 = t * 16;
            f32x8 acc = zf;
#pragma unroll
            for (int c = 0; c < 3; ++c) {
                Frag bf_;
                const uint4* bp = (const uint4*)(Wm + (d0 + lr) * DINP + 32 * c + hi * 8);
                bf_.q[0] = bp[0]; bf_.q[1] = bp[2];
                acc = wmma_bf16(ef[c].v, bf_.v, acc);
            }
#pragma unroll
            for (int r = 0; r < 8; ++r) {
                int row = n0 + r + hi * 8;
                unsigned short hv = f2bf(acc[r]);
                if (mi == 0)      Qb [(size_t)b * NN * DI + (size_t)row * DI + d0 + lr]  = hv;
                else if (mi == 1) Kb [(size_t)b * NN * DI + (size_t)row * DI + d0 + lr]  = hv;
                else              Vtb[(size_t)b * DI * NN + (size_t)(d0 + lr) * NN + row] = hv;
            }
        }
    }
}

// ---------------- kernel 2: per-column (k) online max/sum over all queries
__global__ void __launch_bounds__(32)
col_stats(const unsigned short* __restrict__ Qb, const unsigned short* __restrict__ Kb,
          float* __restrict__ mbuf, float* __restrict__ zbuf) {
    int b = blockIdx.y, kt = blockIdx.x;
    int lane = threadIdx.x, hi = lane >> 4, lr = lane & 15;
    int kbase = kt * 16;
    const unsigned short* Qm = Qb + (size_t)b * NN * DI;
    const unsigned short* Km = Kb + (size_t)b * NN * DI;

    // B-fragments: columns = k indices, inner = DI (fixed for the wave)
    Frag kf[4];
#pragma unroll
    for (int c = 0; c < 4; ++c) {
        const uint4* p = (const uint4*)(Km + (size_t)(kbase + lr) * DI + 32 * c + hi * 8);
        kf[c].q[0] = p[0]; kf[c].q[1] = p[2];
    }
    float run_m = -3.0e38f, run_s = 0.0f;
    f32x8 zf = {};
    for (int qt = 0; qt < NT; ++qt) {
        f32x8 acc = zf;
#pragma unroll
        for (int c = 0; c < 4; ++c) {
            Frag af;
            const uint4* p = (const uint4*)(Qm + (size_t)(qt * 16 + lr) * DI + 32 * c + hi * 8);
            af.q[0] = p[0]; af.q[1] = p[2];      // V0-3 @ +0, V4-7 @ +16 elements
            acc = wmma_bf16(af.v, kf[c].v, acc);
        }
        // lane holds column k=kbase+lr, rows q = qt*16 + r + 8*hi
        float tm = acc[0];
#pragma unroll
        for (int r = 1; r < 8; ++r) tm = fmaxf(tm, acc[r]);
        tm = fmaxf(tm, __shfl_xor(tm, 16, 32));
        float nm = fmaxf(run_m, tm);
        float ts = 0.0f;
#pragma unroll
        for (int r = 0; r < 8; ++r) ts += __expf(acc[r] - nm);
        ts += __shfl_xor(ts, 16, 32);
        run_s = run_s * __expf(run_m - nm) + ts;
        run_m = nm;
    }
    if (hi == 0) {
        mbuf[b * NN + kbase + lr] = run_m;
        zbuf[b * NN + kbase + lr] = 1.0f / run_s;
    }
}

// ---------------- kernel 3: out[q,:] = sum_k exp(s-m_k)*invZ_k * V[k,:]
__global__ void __launch_bounds__(32)
attn_out(const unsigned short* __restrict__ Qb, const unsigned short* __restrict__ Kb,
         const unsigned short* __restrict__ Vtb,
         const float* __restrict__ mbuf, const float* __restrict__ zbuf,
         float* __restrict__ out) {
    int b = blockIdx.y, qt = blockIdx.x;
    int lane = threadIdx.x, hi = lane >> 4, lr = lane & 15;
    int q0 = qt * 16;
    const unsigned short* Qm = Qb + (size_t)b * NN * DI;
    const unsigned short* Km = Kb + (size_t)b * NN * DI;
    const unsigned short* Vm = Vtb + (size_t)b * DI * NN;
    const float* mrow = mbuf + b * NN;
    const float* zrow = zbuf + b * NN;

    // B-fragments of Q^T (columns = q, inner = DI), fixed per wave
    Frag qf[4];
#pragma unroll
    for (int c = 0; c < 4; ++c) {
        const uint4* p = (const uint4*)(Qm + (size_t)(q0 + lr) * DI + 32 * c + hi * 8);
        qf[c].q[0] = p[0]; qf[c].q[1] = p[2];
    }
    f32x8 zf = {};
    f32x8 accO[8];
#pragma unroll
    for (int t = 0; t < 8; ++t) accO[t] = zf;

    for (int kc = 0; kc < NN; kc += 32) {
        // S^T tiles: rows = k (two 16-tiles), cols = q
        f32x8 s0 = zf, s1 = zf;
#pragma unroll
        for (int c = 0; c < 4; ++c) {
            Frag a0, a1;
            const uint4* p0 = (const uint4*)(Km + (size_t)(kc + lr) * DI + 32 * c + hi * 8);
            a0.q[0] = p0[0]; a0.q[1] = p0[2];
            const uint4* p1 = (const uint4*)(Km + (size_t)(kc + 16 + lr) * DI + 32 * c + hi * 8);
            a1.q[0] = p1[0]; a1.q[1] = p1[2];
            s0 = wmma_bf16(a0.v, qf[c].v, s0);
            s1 = wmma_bf16(a1.v, qf[c].v, s1);
        }
        // column-k softmax weights; lane's k indices: kc + hi*8 + r (+16 for tile1)
        const float4* mp0 = (const float4*)(mrow + kc + hi * 8);
        const float4* zp0 = (const float4*)(zrow + kc + hi * 8);
        const float4* mp1 = (const float4*)(mrow + kc + 16 + hi * 8);
        const float4* zp1 = (const float4*)(zrow + kc + 16 + hi * 8);
        float4 ma = mp0[0], mb = mp0[1], za = zp0[0], zb = zp0[1];
        float4 mc = mp1[0], md = mp1[1], zc = zp1[0], zd = zp1[1];
        float m0[8] = {ma.x, ma.y, ma.z, ma.w, mb.x, mb.y, mb.z, mb.w};
        float z0[8] = {za.x, za.y, za.z, za.w, zb.x, zb.y, zb.z, zb.w};
        float m1[8] = {mc.x, mc.y, mc.z, mc.w, md.x, md.y, md.z, md.w};
        float z1[8] = {zc.x, zc.y, zc.z, zc.w, zd.x, zd.y, zd.z, zd.w};

        // P fragment (A-matrix 16x32): lane-local conversion, no shuffles needed
        Frag pf;
#pragma unroll
        for (int i = 0; i < 4; ++i) {
            float e0 = __expf(s0[2*i]     - m0[2*i])     * z0[2*i];
            float e1 = __expf(s0[2*i + 1] - m0[2*i + 1]) * z0[2*i + 1];
            pf.u[i] = pk(e0, e1);
            float e2 = __expf(s1[2*i]     - m1[2*i])     * z1[2*i];
            float e3 = __expf(s1[2*i + 1] - m1[2*i + 1]) * z1[2*i + 1];
            pf.u[4 + i] = pk(e2, e3);
        }
        // O tiles: accO[t] += P(16q x 32k) * V(32k x 16d)
#pragma unroll
        for (int t = 0; t < 8; ++t) {
            Frag vf;
            const uint4* vp = (const uint4*)(Vm + (size_t)(t * 16 + lr) * NN + kc + hi * 8);
            vf.q[0] = vp[0]; vf.q[1] = vp[2];
            accO[t] = wmma_bf16(pf.v, vf.v, accO[t]);
        }
    }
    float* ob = out + (size_t)b * NN * DI;
#pragma unroll
    for (int t = 0; t < 8; ++t)
#pragma unroll
        for (int r = 0; r < 8; ++r)
            ob[(size_t)(q0 + r + hi * 8) * DI + t * 16 + lr] = accO[t][r];
}

// ---------------- host launcher ----------------
extern "C" void kernel_launch(void* const* d_in, const int* in_sizes, int n_in,
                              void* d_out, int out_size, void* d_ws, size_t ws_size,
                              hipStream_t stream) {
    (void)in_sizes; (void)n_in; (void)out_size; (void)ws_size;
    const float* x  = (const float*)d_in[0];
    const float* Wq = (const float*)d_in[1];
    const float* Wk = (const float*)d_in[2];
    const float* Wv = (const float*)d_in[3];

    char* ws = (char*)d_ws;
    const size_t SZ_W  = (size_t)3 * DI * DINP * 2;            // 73,728 B
    const size_t OFF_Q = 0x20000;                              // 128 KB aligned
    const size_t SZ_QK = (size_t)NB * NN * DI * 2;             // 9,437,184 B
    unsigned short* Wpad = (unsigned short*)(ws);
    unsigned short* Qb   = (unsigned short*)(ws + OFF_Q);
    unsigned short* Kb   = (unsigned short*)(ws + OFF_Q + SZ_QK);
    unsigned short* Vtb  = (unsigned short*)(ws + OFF_Q + 2 * SZ_QK);
    float* mbuf          = (float*)(ws + OFF_Q + 3 * SZ_QK);
    float* zbuf          = (float*)(ws + OFF_Q + 3 * SZ_QK + (size_t)NB * NN * 4);
    (void)SZ_W;

    prep_w   <<<(3 * DI * DINP + 255) / 256, 256, 0, stream>>>(Wq, Wk, Wv, Wpad);
    qkv_proj <<<dim3(NT, NB), 32, 0, stream>>>(x, Wpad, Qb, Kb, Vtb);
    col_stats<<<dim3(NT, NB), 32, 0, stream>>>(Qb, Kb, mbuf, zbuf);
    attn_out <<<dim3(NT, NB), 32, 0, stream>>>(Qb, Kb, Vtb, mbuf, zbuf, (float*)d_out);
}
